// TransformerEncoderCell_59519656788390
// MI455X (gfx1250) — matrix operand order repaired and checked by reading, not compiled, not run on hardware
//
#include <hip/hip_runtime.h>

typedef _Float16 f16;
typedef __attribute__((ext_vector_type(16))) _Float16 v16h;
typedef __attribute__((ext_vector_type(8)))  _Float16 v8h;
typedef __attribute__((ext_vector_type(8)))  float    v8f;

// ---------------------------------------------------------------------------
// WMMA helpers (CDNA5 wave32, v_wmma_f32_16x16x32_f16)
// A fragment: 16x32 (MxK) f16. Lane r=lane&15 is row M; lane-half hi=lane>>4.
//   f[j]   (VGPR0-3) = K = hi*8 + j        (j=0..7)
//   f[8+j] (VGPR4-7) = K = 16 + hi*8 + j
// B fragment: 32x16 (KxN) f16. Lane n=lane&15 is col N; hi=lane>>4.
//   f[j] = K = hi*8 + j, f[8+j] = K = 16 + hi*8 + j  (j=0..7)
// D/C:   16x16 f32. d[j]: row M = j + 8*hi, col N = lane&15.
// ---------------------------------------------------------------------------

__device__ __forceinline__ v16h load_a_frag(const f16* __restrict__ base, int ld, int k0) {
  const int lane = threadIdx.x & 31;
  const int r  = lane & 15;
  const int hi = lane >> 4;
  const f16* p = base + r * ld + k0 + (hi << 3);
  v8h lo = *(const v8h*)(p);
  v8h hv = *(const v8h*)(p + 16);
  v16h f;
#pragma unroll
  for (int j = 0; j < 8; ++j) { f[j] = lo[j]; f[8 + j] = hv[j]; }
  return f;
}

// Loads B for D = A x B from a row-major [N x K] source (lane n reads its row).
__device__ __forceinline__ v16h load_b_frag(const f16* __restrict__ base, int ld, int k0) {
  const int lane = threadIdx.x & 31;
  const int n  = lane & 15;
  const int hi = lane >> 4;
  const f16* p = base + n * ld + k0 + (hi << 3);
  v8h lo = *(const v8h*)(p);
  v8h hv = *(const v8h*)(p + 16);
  v16h f;
#pragma unroll
  for (int j = 0; j < 8; ++j) { f[j] = lo[j]; f[8 + j] = hv[j]; }
  return f;
}

__device__ __forceinline__ v8f wmma32(v16h a, v16h b, v8f c) {
  return __builtin_amdgcn_wmma_f32_16x16x32_f16(false, a, false, b, (short)0, c, false, false);
}

// ---------------------------------------------------------------------------
// Problem constants: B=8, L=64, F=16, C=128, H=8, FF=512
//   tokens M = B*L*F = 8192; per-(b,h) positions P = L*F = 1024
// ---------------------------------------------------------------------------
#define NTOK   8192
#define CDIM   128
#define CH     1024
#define PPOS   1024
#define FFDIM  512
#define NBH    64
#define QT     32            // query rows per attention block
#define ATTN_LDS (QT * PPOS * 4 + QT * PPOS * 2)   // 128KB S + 64KB A2 = 192KB

__global__ void cvt_f32_f16_kernel(const float* __restrict__ in, f16* __restrict__ out, int n) {
  int i = blockIdx.x * blockDim.x + threadIdx.x;
  if (i < n) out[i] = (f16)in[i];
}

// QKV projection: y = x @ W^T + b for W in {wq,wk,wv}, scattered to
//   q,k : [bh][p][c]  (f16, row-major, ld=128)
//   vT  : [bh][c][p]  (f16, row-major, ld=1024)
// with head mapping: output feature o = c*8 + h  (reshape (..,128,8) in ref).
__global__ void qkv_kernel(const f16* __restrict__ xh,
                           const f16* __restrict__ wqh, const f16* __restrict__ wkh,
                           const f16* __restrict__ wvh,
                           const float* __restrict__ bq, const float* __restrict__ bk,
                           const float* __restrict__ bv,
                           f16* __restrict__ qh, f16* __restrict__ kh, f16* __restrict__ vT) {
  const int w    = threadIdx.x >> 5;
  const int lane = threadIdx.x & 31;
  const int mt   = blockIdx.x;                // token tile: 0..511
  const int ntg  = blockIdx.y * 4 + w;        // 0..191 (64 n-tiles per matrix x 3)
  const int mat  = ntg >> 6;
  const int nt   = ntg & 63;

  const f16*   W    = (mat == 0) ? wqh : (mat == 1) ? wkh : wvh;
  const float* bias = (mat == 0) ? bq  : (mat == 1) ? bk  : bv;

  const f16* Ab = xh + (size_t)mt * 16 * CDIM;
  const f16* Bb = W  + (size_t)nt * 16 * CDIM;
  v8f acc = {};
#pragma unroll
  for (int kk = 0; kk < 4; ++kk)
    acc = wmma32(load_a_frag(Ab, CDIM, kk * 32), load_b_frag(Bb, CDIM, kk * 32), acc);

  const int col   = lane & 15;
  const int n     = nt * 16 + col;            // 0..1023 within this matrix
  const int h     = n & 7;
  const int cc    = n >> 3;
  const int rbase = (lane >> 4) << 3;
  const float bvv = bias[n];
#pragma unroll
  for (int j = 0; j < 8; ++j) {
    int t  = mt * 16 + rbase + j;
    int b  = t >> 10;
    int p  = t & 1023;
    int bh = b * 8 + h;
    f16 val = (f16)(acc[j] + bvv);
    if (mat == 0)      qh[((size_t)bh * PPOS + p) * CDIM + cc] = val;
    else if (mat == 1) kh[((size_t)bh * PPOS + p) * CDIM + cc] = val;
    else               vT[((size_t)bh * CDIM + cc) * PPOS + p] = val;
  }
}

// Attention: one block per (q-tile of 32 rows, bh). 16 waves (512 threads).
// Dynamic LDS: S (32x1024 f32, 128KB) + A2 (32x1024 f16, 64KB) = 192KB.
__global__ void attn_kernel(const f16* __restrict__ qh, const f16* __restrict__ kh,
                            const f16* __restrict__ vT, f16* __restrict__ zout) {
  extern __shared__ char smem[];
  float* S  = (float*)smem;                      // scores / a1 : [QT][1024]
  f16*   A2 = (f16*)(smem + QT * PPOS * 4);      // final probabilities (f16) : [QT][1024]

  const int w     = threadIdx.x >> 5;            // 0..15
  const int lane  = threadIdx.x & 31;
  const int qt    = blockIdx.x;                  // 0..31
  const int bh    = blockIdx.y;                  // 0..63
  const int col   = lane & 15;
  const int rbase = (lane >> 4) << 3;
  const float scale = 0.08838834764831845f;      // 1/sqrt(128)

  // ---- Phase 1: S = (Q @ K^T) * scale, 32 x 1024 ----
  // wave w handles row-tile rt = w&1, key tiles kt = (w>>1) + 8i.
  const int rt = w & 1;
  const f16* Qb = qh + ((size_t)bh * PPOS + qt * QT + rt * 16) * CDIM;
  v16h aq[4];
#pragma unroll
  for (int kk = 0; kk < 4; ++kk) aq[kk] = load_a_frag(Qb, CDIM, kk * 32);

  for (int kt = (w >> 1); kt < 64; kt += 8) {
    const f16* Kb = kh + ((size_t)bh * PPOS + kt * 16) * CDIM;
    // prefetch next key tile for this wave (global_prefetch_b8 on gfx1250)
    if (kt + 8 < 64) {
      const f16* Kn = Kb + (size_t)8 * 16 * CDIM;
      __builtin_prefetch(Kn + (size_t)(lane & 15) * CDIM + ((lane >> 4) << 6), 0, 3);
    }
    v8f acc = {};
#pragma unroll
    for (int kk = 0; kk < 4; ++kk)
      acc = wmma32(aq[kk], load_b_frag(Kb, CDIM, kk * 32), acc);
#pragma unroll
    for (int j = 0; j < 8; ++j)
      S[(rt * 16 + rbase + j) * PPOS + kt * 16 + col] = acc[j] * scale;
  }
  __syncthreads();

  // ---- Phase 2a: softmax over m (key-L, 64 entries, stride 16) per (r,g) ----
  {
    const int r = threadIdx.x >> 4;              // 0..31 query row
    const int g = threadIdx.x & 15;              // 0..15 key-F
    float* p = S + r * PPOS + g;
    float mx = -3.0e38f;
    for (int m = 0; m < 64; ++m) mx = fmaxf(mx, p[m * 16]);
    float sum = 0.f;
    for (int m = 0; m < 64; ++m) { float e = __expf(p[m * 16] - mx); p[m * 16] = e; sum += e; }
    float inv = 1.f / sum;
    for (int m = 0; m < 64; ++m) p[m * 16] *= inv;
  }
  __syncthreads();

  // ---- Phase 2b: softmax over g (key-F, 16 contiguous) per (r,m) -> A2 f16 ----
#pragma unroll
  for (int ii = 0; ii < 4; ++ii) {
    int rm = threadIdx.x + 512 * ii;             // 0..2047 = (r,m), r=0..31
    int r = rm >> 6, m = rm & 63;
    float* p = S + r * PPOS + m * 16;
    float mx = -3.0e38f;
#pragma unroll
    for (int g = 0; g < 16; ++g) mx = fmaxf(mx, p[g]);
    float e[16], sum = 0.f;
#pragma unroll
    for (int g = 0; g < 16; ++g) { e[g] = __expf(p[g] - mx); sum += e[g]; }
    float inv = 1.f / sum;
    f16* q = A2 + r * PPOS + m * 16;
#pragma unroll
    for (int g = 0; g < 16; ++g) q[g] = (f16)(e[g] * inv);
  }
  __syncthreads();

  // ---- Phase 3: Z = A2 @ V ; wave w -> (row-tile w>>3, channel tile w&7) ----
  const int rt3 = w >> 3;                        // 0..1
  const int n0  = (w & 7) * 16;                  // channel tile
  const f16* Ap = (const f16*)A2 + (size_t)rt3 * 16 * PPOS;
  const f16* Vb = vT + ((size_t)bh * CDIM + n0) * PPOS;   // rows = channels, ld = 1024
  v8f accz = {};
  for (int k0 = 0; k0 < PPOS; k0 += 32)
    accz = wmma32(load_a_frag(Ap, PPOS, k0), load_b_frag(Vb, PPOS, k0), accz);

  const int b = bh >> 3, h = bh & 7;
  const int ocol = (n0 + col) * 8 + h;           // back to feature o = c*8 + h
#pragma unroll
  for (int j = 0; j < 8; ++j) {
    int t = (b << 10) + qt * QT + rt3 * 16 + rbase + j;
    zout[(size_t)t * CH + ocol] = (f16)accz[j];
  }
}

// Out projection (8192x128, K=1024) + residual(x) + LayerNorm1 -> z1 (f32) & z1h (f16)
__global__ void oproj_ln1_kernel(const f16* __restrict__ zout, const f16* __restrict__ woh,
                                 const float* __restrict__ bo, const float* __restrict__ x,
                                 const float* __restrict__ g1, const float* __restrict__ be1,
                                 float* __restrict__ z1, f16* __restrict__ z1h) {
  __shared__ float R[16 * CDIM];
  const int w     = threadIdx.x >> 5;
  const int lane  = threadIdx.x & 31;
  const int mt    = blockIdx.x;                  // 0..511
  const int n0    = w * 16;
  const int col   = lane & 15;
  const int rbase = (lane >> 4) << 3;

  const f16* Ab = zout + (size_t)mt * 16 * CH;
  const f16* Bb = woh + (size_t)n0 * CH;         // wo: [128 x 1024] row-major
  v8f acc = {};
  for (int k0 = 0; k0 < CH; k0 += 32)
    acc = wmma32(load_a_frag(Ab, CH, k0), load_b_frag(Bb, CH, k0), acc);

  const int n = n0 + col;
  const float bias = bo[n];
#pragma unroll
  for (int j = 0; j < 8; ++j) {
    int r = rbase + j;
    int t = mt * 16 + r;
    R[r * CDIM + n] = acc[j] + bias + x[(size_t)t * CDIM + n];
  }
  __syncthreads();

  if (threadIdx.x < 16) {
    int r = threadIdx.x;
    int t = mt * 16 + r;
    float m = 0.f;
    for (int c = 0; c < CDIM; ++c) m += R[r * CDIM + c];
    m *= (1.f / 128.f);
    float v = 0.f;
    for (int c = 0; c < CDIM; ++c) { float d = R[r * CDIM + c] - m; v += d * d; }
    v *= (1.f / 128.f);
    float rs = rsqrtf(v + 1e-5f);
    for (int c = 0; c < CDIM; ++c) {
      float val = (R[r * CDIM + c] - m) * rs * g1[c] + be1[c];
      z1[(size_t)t * CDIM + c]  = val;
      z1h[(size_t)t * CDIM + c] = (f16)val;
    }
  }
}

// FFN1: h1 = relu(z1h @ w1^T + b1), 8192x512, K=128
__global__ void ffn1_kernel(const f16* __restrict__ z1h, const f16* __restrict__ w1h,
                            const float* __restrict__ b1, f16* __restrict__ h1) {
  const int w     = threadIdx.x >> 5;
  const int lane  = threadIdx.x & 31;
  const int mt    = blockIdx.x;                  // 0..511
  const int nt    = blockIdx.y * 8 + w;          // 0..31
  const int n0    = nt * 16;
  const int col   = lane & 15;
  const int rbase = (lane >> 4) << 3;

  const f16* Ab = z1h + (size_t)mt * 16 * CDIM;
  const f16* Bb = w1h + (size_t)n0 * CDIM;       // w1: [512 x 128] row-major
  v8f acc = {};
#pragma unroll
  for (int k0 = 0; k0 < CDIM; k0 += 32)
    acc = wmma32(load_a_frag(Ab, CDIM, k0), load_b_frag(Bb, CDIM, k0), acc);

  const int n = n0 + col;
  const float bias = b1[n];
#pragma unroll
  for (int j = 0; j < 8; ++j) {
    int t = mt * 16 + rbase + j;
    float v = acc[j] + bias;
    h1[(size_t)t * FFDIM + n] = (f16)(v > 0.f ? v : 0.f);
  }
}

// FFN2 (8192x128, K=512) + residual(z1) + LayerNorm2 -> out (f32)
__global__ void ffn2_ln2_kernel(const f16* __restrict__ h1, const f16* __restrict__ w2h,
                                const float* __restrict__ b2, const float* __restrict__ z1,
                                const float* __restrict__ g2, const float* __restrict__ be2,
                                float* __restrict__ out) {
  __shared__ float R[16 * CDIM];
  const int w     = threadIdx.x >> 5;
  const int lane  = threadIdx.x & 31;
  const int mt    = blockIdx.x;                  // 0..511
  const int n0    = w * 16;
  const int col   = lane & 15;
  const int rbase = (lane >> 4) << 3;

  const f16* Ab = h1 + (size_t)mt * 16 * FFDIM;
  const f16* Bb = w2h + (size_t)n0 * FFDIM;      // w2: [128 x 512] row-major
  v8f acc = {};
  for (int k0 = 0; k0 < FFDIM; k0 += 32)
    acc = wmma32(load_a_frag(Ab, FFDIM, k0), load_b_frag(Bb, FFDIM, k0), acc);

  const int n = n0 + col;
  const float bias = b2[n];
#pragma unroll
  for (int j = 0; j < 8; ++j) {
    int r = rbase + j;
    int t = mt * 16 + r;
    R[r * CDIM + n] = acc[j] + bias + z1[(size_t)t * CDIM + n];
  }
  __syncthreads();

  if (threadIdx.x < 16) {
    int r = threadIdx.x;
    int t = mt * 16 + r;
    float m = 0.f;
    for (int c = 0; c < CDIM; ++c) m += R[r * CDIM + c];
    m *= (1.f / 128.f);
    float v = 0.f;
    for (int c = 0; c < CDIM; ++c) { float d = R[r * CDIM + c] - m; v += d * d; }
    v *= (1.f / 128.f);
    float rs = rsqrtf(v + 1e-5f);
    for (int c = 0; c < CDIM; ++c)
      out[(size_t)t * CDIM + c] = (R[r * CDIM + c] - m) * rs * g2[c] + be2[c];
  }
}

extern "C" void kernel_launch(void* const* d_in, const int* in_sizes, int n_in,
                              void* d_out, int out_size, void* d_ws, size_t ws_size,
                              hipStream_t stream) {
  const float* x    = (const float*)d_in[0];
  const float* wq   = (const float*)d_in[1];
  const float* bq   = (const float*)d_in[2];
  const float* wk   = (const float*)d_in[3];
  const float* bk   = (const float*)d_in[4];
  const float* wv   = (const float*)d_in[5];
  const float* bv   = (const float*)d_in[6];
  const float* wo   = (const float*)d_in[7];
  const float* bo   = (const float*)d_in[8];
  const float* ln1g = (const float*)d_in[9];
  const float* ln1b = (const float*)d_in[10];
  const float* w1   = (const float*)d_in[11];
  const float* b1   = (const float*)d_in[12];
  const float* w2   = (const float*)d_in[13];
  const float* b2   = (const float*)d_in[14];
  const float* ln2g = (const float*)d_in[15];
  const float* ln2b = (const float*)d_in[16];
  float* out = (float*)d_out;

  // Workspace bump allocator (256B aligned)
  char* ws = (char*)d_ws;
  size_t off = 0;
  auto alloc = [&](size_t bytes) -> void* {
    void* p = ws + off;
    off = (off + bytes + 255) & ~(size_t)255;
    return p;
  };
  f16* xh   = (f16*)alloc((size_t)NTOK * CDIM * 2);
  f16* wqh  = (f16*)alloc((size_t)CH * CDIM * 2);
  f16* wkh  = (f16*)alloc((size_t)CH * CDIM * 2);
  f16* wvh  = (f16*)alloc((size_t)CH * CDIM * 2);
  f16* woh  = (f16*)alloc((size_t)CDIM * CH * 2);
  f16* w1h  = (f16*)alloc((size_t)FFDIM * CDIM * 2);
  f16* w2h  = (f16*)alloc((size_t)CDIM * FFDIM * 2);
  f16* qh   = (f16*)alloc((size_t)NBH * PPOS * CDIM * 2);
  f16* kh   = (f16*)alloc((size_t)NBH * PPOS * CDIM * 2);
  f16* vT   = (f16*)alloc((size_t)NBH * CDIM * PPOS * 2);
  f16* zout = (f16*)alloc((size_t)NTOK * CH * 2);
  float* z1 = (float*)alloc((size_t)NTOK * CDIM * 4);
  f16* z1h  = (f16*)alloc((size_t)NTOK * CDIM * 2);
  f16* h1   = (f16*)alloc((size_t)NTOK * FFDIM * 2);

  auto cvt = [&](const float* in, f16* o, int n) {
    cvt_f32_f16_kernel<<<(n + 255) / 256, 256, 0, stream>>>(in, o, n);
  };
  cvt(x,  xh,  NTOK * CDIM);
  cvt(wq, wqh, CH * CDIM);
  cvt(wk, wkh, CH * CDIM);
  cvt(wv, wvh, CH * CDIM);
  cvt(wo, woh, CDIM * CH);
  cvt(w1, w1h, FFDIM * CDIM);
  cvt(w2, w2h, CDIM * FFDIM);

  qkv_kernel<<<dim3(512, 48), 128, 0, stream>>>(xh, wqh, wkh, wvh, bq, bk, bv, qh, kh, vT);

  (void)hipFuncSetAttribute((const void*)attn_kernel,
                            hipFuncAttributeMaxDynamicSharedMemorySize, ATTN_LDS);
  attn_kernel<<<dim3(PPOS / QT, NBH), 512, ATTN_LDS, stream>>>(qh, kh, vT, zout);

  oproj_ln1_kernel<<<512, 256, 0, stream>>>(zout, woh, bo, x, ln1g, ln1b, z1, z1h);
  ffn1_kernel<<<dim3(512, 4), 256, 0, stream>>>(z1h, w1h, b1, h1);
  ffn2_ln2_kernel<<<512, 256, 0, stream>>>(h1, w2h, b2, z1, ln2g, ln2b, out);

  (void)in_sizes; (void)n_in; (void)out_size; (void)ws_size;
}
